// DLRMNet_5677946766095
// MI455X (gfx1250) — matrix operand order repaired and checked by reading, not compiled, not run on hardware
//
#include <hip/hip_runtime.h>
#include <hip/hip_bf16.h>
#include <math.h>

#define TPB 256

typedef __attribute__((ext_vector_type(16))) __bf16 v16bf;
typedef __attribute__((ext_vector_type(8)))  float  v8f;

struct __attribute__((aligned(16))) U4 { unsigned int x, y, z, w; };
union Frag { v16bf v; U4 q[2]; unsigned short s[16]; };

__device__ __forceinline__ unsigned short f2b(float f) {
  union { float f; unsigned int u; } c; c.f = f;
  return (unsigned short)((c.u + 0x7FFFu + ((c.u >> 16) & 1u)) >> 16);
}
__device__ __forceinline__ float b2f(unsigned short h) {
  union { float f; unsigned int u; } c; c.u = ((unsigned int)h) << 16;
  return c.f;
}

// ---------------- bottom MLP: 13 -> 512 -> 256 -> 64 -> 16, ReLU each ----------------
__device__ void mlp_layer(const float* __restrict__ in, int K,
                          float* out, int N,
                          const float* __restrict__ w, const float* __restrict__ b,
                          int tid) {
  for (int idx = tid; idx < 16 * N; idx += TPB) {
    int r = idx / N, c = idx - r * N;
    float acc = b[c];
    const float* xi = in + r * K;
    const float* wc = w + c;
    for (int k = 0; k < K; ++k) acc = fmaf(xi[k], wc[(size_t)k * N], acc);
    out[idx] = fmaxf(acc, 0.0f);
  }
}

__global__ void __launch_bounds__(TPB) dlrm_bottom_mlp(
    const float* __restrict__ dense,
    const float* __restrict__ bw0, const float* __restrict__ bb0,
    const float* __restrict__ bw1, const float* __restrict__ bb1,
    const float* __restrict__ bw2, const float* __restrict__ bb2,
    const float* __restrict__ bw3, const float* __restrict__ bb3,
    float* __restrict__ xbuf) {
  __shared__ float hA[16 * 512];
  __shared__ float hB[16 * 512];
  int tid = threadIdx.x;
  int r0 = blockIdx.x * 16;
  for (int idx = tid; idx < 16 * 13; idx += TPB)
    hA[idx] = dense[r0 * 13 + idx];
  __syncthreads();
  mlp_layer(hA, 13, hB, 512, bw0, bb0, tid); __syncthreads();
  mlp_layer(hB, 512, hA, 256, bw1, bb1, tid); __syncthreads();
  mlp_layer(hA, 256, hB, 64, bw2, bb2, tid); __syncthreads();
  mlp_layer(hB, 64, hA, 16, bw3, bb3, tid); __syncthreads();
  {
    int r = tid >> 4, c = tid & 15;
    xbuf[(size_t)(r0 + r) * 432 + c] = hA[tid];
  }
}

// ---------------- embedding gather into x[:,16:432] ----------------
__global__ void __launch_bounds__(TPB) dlrm_gather(
    const int* __restrict__ sparse, const float* __restrict__ emb,
    float* __restrict__ xbuf) {
  int g = blockIdx.x * TPB + threadIdx.x;
  if (g >= 1024 * 26) return;
  int b = g / 26, f = g - b * 26;
  int idx = sparse[g];
  const U4* src = (const U4*)(emb + ((size_t)f * 100000 + (size_t)idx) * 16);
  U4* dst = (U4*)(xbuf + (size_t)b * 432 + 16 + f * 16);
  dst[0] = src[0]; dst[1] = src[1]; dst[2] = src[2]; dst[3] = src[3];
}

// ---------------- interaction: z = [dense16 | triu(x x^T) | pad24], bf16 ----------------
__global__ void __launch_bounds__(TPB) dlrm_interact(
    const float* __restrict__ xbuf, unsigned short* __restrict__ z) {
  __shared__ float xr[432];
  int b = blockIdx.x, tid = threadIdx.x;
  for (int i = tid; i < 432; i += TPB) xr[i] = xbuf[(size_t)b * 432 + i];
  __syncthreads();
  unsigned short* zr = z + (size_t)b * 93568;
  for (int idx = tid; idx < 93568; idx += TPB) {
    float val;
    if (idx < 16) {
      val = xr[idx];
    } else if (idx < 93544) {
      int p = idx - 16;
      float disc = 748225.0f - 8.0f * (float)p;   // (2*432+1)^2
      int i = (int)((865.0f - sqrtf(disc)) * 0.5f);
      if (i < 0) i = 0;
      if (i > 431) i = 431;
      while (i > 0 && (i * 432 - (i * (i - 1)) / 2) > p) --i;
      while (i < 431 && ((i + 1) * 432 - ((i + 1) * i) / 2) <= p) ++i;
      int j = i + (p - (i * 432 - (i * (i - 1)) / 2));
      val = xr[i] * xr[j];
    } else {
      val = 0.0f;  // K padding 93544 -> 93568
    }
    zr[idx] = f2b(val);
  }
}

// ---------------- fp32 -> bf16 transpose-convert: in (Ksrc x N) -> out (N x Kp), pad K ----
// 32x32 LDS-tiled; both global read and write coalesced. Requires N%32==0, Kp%32==0.
__global__ void __launch_bounds__(TPB) dlrm_cvtT(
    const float* __restrict__ in, unsigned short* __restrict__ out,
    int Ksrc, int N, int Kp) {
  __shared__ float t[32][33];
  int k0 = blockIdx.x * 32, n0 = blockIdx.y * 32;
  int tid = threadIdx.x;
  int c = tid & 31, rb = tid >> 5;       // 8 rows per pass, 4 passes
#pragma unroll
  for (int i = 0; i < 4; ++i) {
    int r = rb + i * 8;                  // k within tile
    int gk = k0 + r;
    t[r][c] = (gk < Ksrc) ? in[(size_t)gk * N + n0 + c] : 0.0f;
  }
  __syncthreads();
#pragma unroll
  for (int i = 0; i < 4; ++i) {
    int r = rb + i * 8;                  // n within tile
    out[(size_t)(n0 + r) * Kp + k0 + c] = f2b(t[c][r]);
  }
}

// ---------------- bf16 WMMA GEMM, double-buffered LDS pipeline ----------------
// A: MxK bf16 row-major. Wt: NxK bf16 row-major (pre-transposed weights).
// out = act(A @ Wt^T + bias). M%128==0, N%128==0, K%32==0.
__global__ void __launch_bounds__(TPB) dlrm_wmma_gemm(
    const unsigned short* __restrict__ A,
    const unsigned short* __restrict__ Wt,
    const float* __restrict__ bias,
    unsigned short* __restrict__ outB,
    int M, int N, int K, int relu) {
  __shared__ unsigned short As[2][128][40];  // [buf][m][k], 80B rows -> 16B aligned
  __shared__ unsigned short Bs[2][128][40];  // [buf][n][k]
  int tid = threadIdx.x;
  int lane = tid & 31, wv = tid >> 5;
  int ln = lane & 15, lh = lane >> 4;
  int wm = (wv & 3) * 32, wn = (wv >> 2) * 64;  // 4x2 wave grid, 32x64 wave tile
  int bm = blockIdx.y * 128, bn = blockIdx.x * 128;

  int lr = tid >> 1, lc = (tid & 1) << 4;  // loader: 2 thr/row, 16 bf16 each (A and Wt)

  v8f acc[2][4];
#pragma unroll
  for (int i = 0; i < 2; ++i)
#pragma unroll
    for (int j = 0; j < 4; ++j)
#pragma unroll
      for (int e = 0; e < 8; ++e) acc[i][j][e] = 0.0f;

  const int nk = K >> 5;
  U4 ra0, ra1, rb0, rb1;

  // prologue: fetch tile 0 and stage into buffer 0
  {
    const U4* as = (const U4*)(A  + (size_t)(bm + lr) * K + lc);
    ra0 = as[0]; ra1 = as[1];
    const U4* bs = (const U4*)(Wt + (size_t)(bn + lr) * K + lc);
    rb0 = bs[0]; rb1 = bs[1];
    U4* da = (U4*)&As[0][lr][lc]; da[0] = ra0; da[1] = ra1;
    U4* db = (U4*)&Bs[0][lr][lc]; db[0] = rb0; db[1] = rb1;
  }

  for (int kt = 0; kt < nk; ++kt) {
    __syncthreads();
    int buf = kt & 1;
    // issue global loads for next tile before computing current one
    if (kt + 1 < nk) {
      int k0 = (kt + 1) << 5;
      const U4* as = (const U4*)(A  + (size_t)(bm + lr) * K + k0 + lc);
      ra0 = as[0]; ra1 = as[1];
      const U4* bs = (const U4*)(Wt + (size_t)(bn + lr) * K + k0 + lc);
      rb0 = bs[0]; rb1 = bs[1];
    }
    if (kt + 2 < nk) {  // global_prefetch_b8 two tiles ahead
      int kp = (kt + 2) << 5;
      __builtin_prefetch(A  + (size_t)(bm + lr) * K + kp + lc, 0, 1);
      __builtin_prefetch(Wt + (size_t)(bn + lr) * K + kp + lc, 0, 1);
    }

    Frag af[2], bfr[4];
#pragma unroll
    for (int mt = 0; mt < 2; ++mt) {  // A frag: e<8 -> K=lh*8+e ; e>=8 -> K=16+lh*8+(e-8)
      int row = wm + mt * 16 + ln;
      af[mt].q[0] = *(const U4*)&As[buf][row][lh * 8];
      af[mt].q[1] = *(const U4*)&As[buf][row][16 + lh * 8];
    }
#pragma unroll
    for (int nt = 0; nt < 4; ++nt) {  // B frag: lane=N, e<8 -> K=lh*8+e ; e>=8 -> K=16+lh*8+(e-8)
      int col = wn + nt * 16 + ln;
      bfr[nt].q[0] = *(const U4*)&Bs[buf][col][lh * 8];
      bfr[nt].q[1] = *(const U4*)&Bs[buf][col][16 + lh * 8];
    }
#pragma unroll
    for (int mt = 0; mt < 2; ++mt)
#pragma unroll
      for (int nt = 0; nt < 4; ++nt)
        acc[mt][nt] = __builtin_amdgcn_wmma_f32_16x16x32_bf16(
            false, af[mt].v, false, bfr[nt].v, (short)0, acc[mt][nt], false, false);

    // stage next tile into the other buffer (safe: readers of it synced 1 iter ago)
    if (kt + 1 < nk) {
      int nb = (kt + 1) & 1;
      U4* da = (U4*)&As[nb][lr][lc]; da[0] = ra0; da[1] = ra1;
      U4* db = (U4*)&Bs[nb][lr][lc]; db[0] = rb0; db[1] = rb1;
    }
  }

  // epilogue: C/D layout -> M = lh*8 + r, N = ln
#pragma unroll
  for (int mt = 0; mt < 2; ++mt)
#pragma unroll
    for (int nt = 0; nt < 4; ++nt) {
      int gn = bn + wn + nt * 16 + ln;
      float bv = bias[gn];
#pragma unroll
      for (int r = 0; r < 8; ++r) {
        int gm = bm + wm + mt * 16 + lh * 8 + r;
        float v = acc[mt][nt][r] + bv;
        if (relu) v = fmaxf(v, 0.0f);
        outB[(size_t)gm * N + gn] = f2b(v);
      }
    }
}

// ---------------- final layer: 256 -> 1, no ReLU ----------------
__global__ void __launch_bounds__(TPB) dlrm_final(
    const unsigned short* __restrict__ a3, const float* __restrict__ tw3,
    const float* __restrict__ tb3, float* __restrict__ out) {
  int b = blockIdx.x * TPB + threadIdx.x;
  if (b >= 1024) return;
  float acc = tb3[0];
  const unsigned short* row = a3 + (size_t)b * 256;
  for (int k = 0; k < 256; ++k) acc = fmaf(b2f(row[k]), tw3[k], acc);
  out[b] = acc;
}

extern "C" void kernel_launch(void* const* d_in, const int* in_sizes, int n_in,
                              void* d_out, int out_size, void* d_ws, size_t ws_size,
                              hipStream_t stream) {
  (void)in_sizes; (void)n_in; (void)out_size; (void)ws_size;
  const float* dense = (const float*)d_in[0];
  const int*   sparse = (const int*)d_in[1];
  const float* emb  = (const float*)d_in[2];
  const float* bw0 = (const float*)d_in[3];  const float* bb0 = (const float*)d_in[4];
  const float* bw1 = (const float*)d_in[5];  const float* bb1 = (const float*)d_in[6];
  const float* bw2 = (const float*)d_in[7];  const float* bb2 = (const float*)d_in[8];
  const float* bw3 = (const float*)d_in[9];  const float* bb3 = (const float*)d_in[10];
  const float* tw0 = (const float*)d_in[11]; const float* tb0 = (const float*)d_in[12];
  const float* tw1 = (const float*)d_in[13]; const float* tb1 = (const float*)d_in[14];
  const float* tw2 = (const float*)d_in[15]; const float* tb2 = (const float*)d_in[16];
  const float* tw3 = (const float*)d_in[17]; const float* tb3 = (const float*)d_in[18];
  float* out = (float*)d_out;

  char* ws = (char*)d_ws;
  size_t off = 0;
  auto take = [&](size_t bytes) -> void* {
    void* p = (void*)(ws + off);
    off += (bytes + 255) & ~(size_t)255;
    return p;
  };
  float*          xbuf  = (float*)take((size_t)1024 * 432 * 4);
  unsigned short* zb    = (unsigned short*)take((size_t)1024 * 93568 * 2);
  unsigned short* tw0t  = (unsigned short*)take((size_t)1024 * 93568 * 2);  // N x Kp
  unsigned short* tw1t  = (unsigned short*)take((size_t)512 * 1024 * 2);
  unsigned short* tw2t  = (unsigned short*)take((size_t)256 * 512 * 2);
  unsigned short* a1b   = (unsigned short*)take((size_t)1024 * 1024 * 2);
  unsigned short* a2b   = (unsigned short*)take((size_t)1024 * 512 * 2);
  unsigned short* a3b   = (unsigned short*)take((size_t)1024 * 256 * 2);

  dlrm_bottom_mlp<<<dim3(64), dim3(TPB), 0, stream>>>(
      dense, bw0, bb0, bw1, bb1, bw2, bb2, bw3, bb3, xbuf);
  dlrm_gather<<<dim3((1024 * 26 + TPB - 1) / TPB), dim3(TPB), 0, stream>>>(sparse, emb, xbuf);
  dlrm_interact<<<dim3(1024), dim3(TPB), 0, stream>>>(xbuf, zb);

  // transpose-convert weights to bf16 NxK (handles K zero-padding for tw0)
  dlrm_cvtT<<<dim3(93568 / 32, 1024 / 32), dim3(TPB), 0, stream>>>(tw0, tw0t, 93544, 1024, 93568);
  dlrm_cvtT<<<dim3(1024 / 32, 512 / 32),   dim3(TPB), 0, stream>>>(tw1, tw1t, 1024, 512, 1024);
  dlrm_cvtT<<<dim3(512 / 32, 256 / 32),    dim3(TPB), 0, stream>>>(tw2, tw2t, 512, 256, 512);

  // top MLP
  dlrm_wmma_gemm<<<dim3(8, 8), dim3(TPB), 0, stream>>>(zb,  tw0t, tb0, a1b, 1024, 1024, 93568, 1);
  dlrm_wmma_gemm<<<dim3(4, 8), dim3(TPB), 0, stream>>>(a1b, tw1t, tb1, a2b, 1024, 512, 1024, 1);
  dlrm_wmma_gemm<<<dim3(2, 8), dim3(TPB), 0, stream>>>(a2b, tw2t, tb2, a3b, 1024, 256, 512, 1);
  dlrm_final<<<dim3(4), dim3(TPB), 0, stream>>>(a3b, tw3, tb3, out);
}
